// FeatureFusionNetwork_17738214933105
// MI455X (gfx1250) — matrix operand order repaired and checked by reading, not compiled, not run on hardware
//
#include <hip/hip_runtime.h>
#include <hip/hip_bf16.h>

#define D_MODEL 512
#define DIM_FF  256
#define NTOK    3072
#define BATCH   8
#define TOPK    256

typedef __bf16 bf16_t;
typedef __attribute__((ext_vector_type(16))) __bf16 v16bf;
typedef __attribute__((ext_vector_type(8)))  __bf16 v8bf;
typedef __attribute__((ext_vector_type(8)))  float  v8f;

// ---------------------------------------------------------------------------
// WMMA fragment loaders (bf16 16x16x32, ISA 7.12.2 layouts, wave32)
// A (16Mx32K): lanes 0-15 -> row=lane, K {k0..k0+7, k0+16..k0+23}
//              lanes16-31 -> row=lane-16, K {k0+8..k0+15, k0+24..k0+31}
// B (32Kx16N): lane -> col = lane%16, K same as A (table order)
// ---------------------------------------------------------------------------

__device__ __forceinline__ v16bf fragA_from_f32(const float* __restrict__ base,
                                                int ld, int row0, int k0) {
  int lane = threadIdx.x & 31;
  int half = lane >> 4;
  const float* p = base + (size_t)(row0 + (lane & 15)) * ld + k0 + half * 8;
  v16bf f;
#pragma unroll
  for (int i = 0; i < 8; ++i) f[i] = (bf16_t)p[i];
#pragma unroll
  for (int i = 0; i < 8; ++i) f[8 + i] = (bf16_t)p[16 + i];
  return f;
}

__device__ __forceinline__ v16bf fragB_from_f32(const float* __restrict__ base,
                                                int ld, int col0, int k0) {
  int lane = threadIdx.x & 31;
  const float* p = base + (size_t)(col0 + (lane & 15)) * ld + k0 + (lane >> 4) * 8;
  v16bf f;
#pragma unroll
  for (int i = 0; i < 16; ++i) f[i] = (bf16_t)p[i < 8 ? i : i + 8];
  return f;
}

__device__ __forceinline__ v16bf fragA_bf16(const bf16_t* __restrict__ base,
                                            int ld, int row0, int k0) {
  int lane = threadIdx.x & 31;
  int half = lane >> 4;
  const bf16_t* p = base + (size_t)(row0 + (lane & 15)) * ld + k0 + half * 8;
  v8bf lo = *(const v8bf*)p;         // 16B load
  v8bf hi = *(const v8bf*)(p + 16);  // 16B load
  v16bf f;
#pragma unroll
  for (int i = 0; i < 8; ++i) { f[i] = lo[i]; f[8 + i] = hi[i]; }
  return f;
}

__device__ __forceinline__ v16bf fragB_bf16(const bf16_t* __restrict__ base,
                                            int ld, int col0, int k0) {
  int lane = threadIdx.x & 31;
  const bf16_t* p = base + (size_t)(col0 + (lane & 15)) * ld + k0 + (lane >> 4) * 8;
  v8bf lo = *(const v8bf*)p;
  v8bf hi = *(const v8bf*)(p + 16);
  v16bf f;
#pragma unroll
  for (int i = 0; i < 8; ++i) { f[i] = lo[i]; f[8 + i] = hi[i]; }
  return f;
}

// Monotone key: larger float -> larger unsigned key
__device__ __forceinline__ unsigned fkey(float f) {
  unsigned u = __float_as_uint(f);
  return u ^ ((u >> 31) ? 0xFFFFFFFFu : 0x80000000u);
}

// ---------------------------------------------------------------------------
// Kernel 1: Q = X*W1^T + b1 ; K = X*W2^T + b2  (f32 in, bf16 out)
// Block = 16 rows of X, 8 waves x 4 col-tiles (32 tiles: 16 Q + 16 K)
// ---------------------------------------------------------------------------
__global__ void __launch_bounds__(256) qk_proj_kernel(
    const float* __restrict__ X,
    const float* __restrict__ W1, const float* __restrict__ b1,
    const float* __restrict__ W2, const float* __restrict__ b2,
    bf16_t* __restrict__ Q, bf16_t* __restrict__ Km) {
  int rowBase = blockIdx.x * 16;
  int wid  = threadIdx.x >> 5;
  int lane = threadIdx.x & 31;

  // Preload all A fragments for this row tile (K = 512 -> 16 chunks of 32)
  v16bf aF[16];
#pragma unroll
  for (int kc = 0; kc < 16; ++kc)
    aF[kc] = fragA_from_f32(X, D_MODEL, rowBase, kc * 32);

  for (int t = wid * 4; t < wid * 4 + 4; ++t) {   // wave-uniform
    const float* W    = (t < 16) ? W1 : W2;
    const float* bias = (t < 16) ? b1 : b2;
    bf16_t*      dst  = (t < 16) ? Q  : Km;
    int colBase = (t & 15) * 16;

    v8f acc = {};
#pragma unroll
    for (int kc = 0; kc < 16; ++kc) {
      v16bf bF = fragB_from_f32(W, D_MODEL, colBase, kc * 32);
      acc = __builtin_amdgcn_wmma_f32_16x16x32_bf16(
          false, aF[kc], false, bF, (short)0, acc, false, false);
    }
    float bv = bias[colBase + (lane & 15)];
#pragma unroll
    for (int r = 0; r < 8; ++r) {
      int M = r + (lane >> 4) * 8;
      dst[(size_t)(rowBase + M) * DIM_FF + colBase + (lane & 15)] =
          (bf16_t)(acc[r] + bv);
    }
  }
}

// ---------------------------------------------------------------------------
// Kernel 2: S = Q*K^T per batch (bf16 WMMA, f32 acc), exact top-256 per row
// via 4-round radix select on float keys, scatter 0.5 into out[n,m]+out[m,n].
// Block = (batch, 16-row tile). 192 KB dynamic LDS of scores (CDNA5: 320 KB).
// ---------------------------------------------------------------------------
__global__ void __launch_bounds__(256) attn_topk_kernel(
    const bf16_t* __restrict__ Q, const bf16_t* __restrict__ Km,
    float* __restrict__ out) {
  __shared__ unsigned hist[8][256];
  __shared__ unsigned bc[8][2];
  __shared__ unsigned tie[8];
  extern __shared__ float scores[];  // [16][NTOK]

  int blk = blockIdx.x;
  int b   = blk / (NTOK / 16);
  int rt  = blk % (NTOK / 16);
  int rowBase = rt * 16;
  int wid  = threadIdx.x >> 5;
  int lane = threadIdx.x & 31;

  const bf16_t* Qb = Q  + (size_t)b * NTOK * DIM_FF;
  const bf16_t* Kb = Km + (size_t)b * NTOK * DIM_FF;

  // Q fragments for this 16-row tile, K = 256 -> 8 chunks
  v16bf qF[8];
#pragma unroll
  for (int kc = 0; kc < 8; ++kc)
    qF[kc] = fragA_bf16(Qb, DIM_FF, rowBase, kc * 32);

  // 192 column tiles split over 8 waves
  for (int ct = wid; ct < NTOK / 16; ct += 8) {   // wave-uniform
    v8f acc = {};
#pragma unroll
    for (int kc = 0; kc < 8; ++kc) {
      v16bf bF = fragB_bf16(Kb, DIM_FF, ct * 16, kc * 32);
      acc = __builtin_amdgcn_wmma_f32_16x16x32_bf16(
          false, qF[kc], false, bF, (short)0, acc, false, false);
    }
#pragma unroll
    for (int r = 0; r < 8; ++r) {
      int M = r + (lane >> 4) * 8;
      scores[M * NTOK + ct * 16 + (lane & 15)] = acc[r];
    }
  }
  __syncthreads();

  // Exact top-TOPK per row; wave w handles rows 2w and 2w+1 (uniform barriers)
  float* outb = out + (size_t)b * NTOK * NTOK;
  for (int rp = 0; rp < 2; ++rp) {
    int row = wid * 2 + rp;
    const float* srow = scores + row * NTOK;

    unsigned prefix = 0, pmask = 0, remaining = TOPK;
    for (int round = 0; round < 4; ++round) {
      int shift = 24 - 8 * round;
#pragma unroll
      for (int bb = lane; bb < 256; bb += 32) hist[wid][bb] = 0;
      __syncthreads();
      for (int j = 0; j < NTOK / 32; ++j) {
        unsigned key = fkey(srow[lane + 32 * j]);
        if ((key & pmask) == prefix)
          atomicAdd(&hist[wid][(key >> shift) & 0xFF], 1u);
      }
      __syncthreads();
      // descending cumulative: lane L owns bins [8L, 8L+7]
      unsigned h[8]; unsigned lsum = 0;
#pragma unroll
      for (int t2 = 0; t2 < 8; ++t2) { h[t2] = hist[wid][lane * 8 + t2]; lsum += h[t2]; }
      unsigned suff = lsum;
#pragma unroll
      for (int d = 1; d < 32; d <<= 1) {
        unsigned o = __shfl_down(suff, d, 32);
        if (lane + d < 32) suff += o;
      }
      unsigned cum = suff - lsum;  // count in bins above my range
#pragma unroll
      for (int t2 = 7; t2 >= 0; --t2) {
        unsigned nc = cum + h[t2];
        if (cum < remaining && nc >= remaining) {
          bc[wid][0] = prefix | ((unsigned)(lane * 8 + t2) << shift);
          bc[wid][1] = remaining - cum;
        }
        cum = nc;
      }
      __syncthreads();
      prefix    = bc[wid][0];
      remaining = bc[wid][1];
      pmask    |= 0xFFu << shift;
    }

    if (lane == 0) tie[wid] = 0;
    __syncthreads();

    unsigned T = prefix;           // exact key of the TOPK-th largest
    int nglob = rowBase + row;
    for (int j = 0; j < NTOK / 32; ++j) {
      int m = lane + 32 * j;
      unsigned key = fkey(srow[m]);
      bool sel = key > T;
      if (key == T) {
        unsigned old = atomicAdd(&tie[wid], 1u);
        sel = old < remaining;     // include exactly `remaining` ties
      }
      if (sel) {
        atomicAdd(outb + (size_t)nglob * NTOK + m, 0.5f);
        atomicAdd(outb + (size_t)m * NTOK + nglob, 0.5f);
      }
    }
    __syncthreads();
  }
}

// ---------------------------------------------------------------------------
extern "C" void kernel_launch(void* const* d_in, const int* in_sizes, int n_in,
                              void* d_out, int out_size, void* d_ws, size_t ws_size,
                              hipStream_t stream) {
  (void)in_sizes; (void)n_in; (void)ws_size;
  const float* src = (const float*)d_in[0];   // [B, N, 512]
  const float* W1  = (const float*)d_in[1];   // [256, 512]
  const float* b1  = (const float*)d_in[2];
  const float* W2  = (const float*)d_in[3];
  const float* b2  = (const float*)d_in[4];
  float* out = (float*)d_out;                 // [B, N, N]

  bf16_t* Q  = (bf16_t*)d_ws;
  bf16_t* Km = (bf16_t*)((char*)d_ws +
               (size_t)BATCH * NTOK * DIM_FF * sizeof(bf16_t));

  // Output is accumulated with atomics -> zero it every launch (capture-safe).
  hipMemsetAsync(d_out, 0, (size_t)out_size * sizeof(float), stream);

  qk_proj_kernel<<<(BATCH * NTOK) / 16, 256, 0, stream>>>(src, W1, b1, W2, b2, Q, Km);

  size_t smem = (size_t)16 * NTOK * sizeof(float);  // 192 KB (CDNA5 WGP: 320 KB)
  hipFuncSetAttribute((const void*)attn_topk_kernel,
                      hipFuncAttributeMaxDynamicSharedMemorySize, (int)smem);
  attn_topk_kernel<<<BATCH * (NTOK / 16), 256, smem, stream>>>(Q, Km, out);
}
